// Classifier_2181843386986
// MI455X (gfx1250) — matrix operand order repaired and checked by reading, not compiled, not run on hardware
//
#include <hip/hip_runtime.h>

// ---------------- types / constants ----------------
typedef unsigned short u16;
typedef __attribute__((ext_vector_type(8)))  __bf16 v8bf;
typedef __attribute__((ext_vector_type(16))) __bf16 v16bf;
typedef __attribute__((ext_vector_type(8)))  float  v8f;
typedef __attribute__((ext_vector_type(4)))  unsigned v4u;
typedef __attribute__((ext_vector_type(8)))  int      v8i;
typedef __attribute__((ext_vector_type(4)))  int      v4i;

#define V_   50000
#define D_   300
#define DP_  320      // D padded to multiple of 32 (K dim for WMMA)
#define B_   32
#define S_   128
#define H_   1024
#define NT_  9
#define TOK  (B_*S_)  // 4096 tokens, row index = s*32 + b  (seq-major, matches transpose(1,0,2))
#define G4H  (4*H_)   // 4096 gate columns
#define H2_  (2*H_)   // 2048

// ---------------- helpers ----------------
__device__ inline u16 f2bf(float f) {
  unsigned u = __builtin_bit_cast(unsigned, f);
  unsigned r = (u + 0x7FFFu + ((u >> 16) & 1u)) >> 16;   // RNE
  return (u16)r;
}
__device__ inline float bf2f(u16 h) {
  unsigned u = ((unsigned)h) << 16;
  return __builtin_bit_cast(float, u);
}
__device__ inline v8f wmma_bf16(v16bf a, v16bf b, v8f c) {
  // D = A(16x32) * B(32x16) + C ; f32 accumulate
  return __builtin_amdgcn_wmma_f32_16x16x32_bf16(false, a, false, b, (short)0, c, false, false);
}
// A fragment per ISA layout: lane(r,half): row r, K = {half*8..+7} U {16+half*8..+7}
__device__ inline v16bf ld_afrag(const u16* rowbase, int kb, int half) {
  v8bf lo = *(const v8bf*)(rowbase + kb + half * 8);
  v8bf hi = *(const v8bf*)(rowbase + kb + 16 + half * 8);
  return __builtin_shufflevector(lo, hi, 0,1,2,3,4,5,6,7,8,9,10,11,12,13,14,15);
}
// B fragment: lane(r,half): column r of B == row r of W (N x K row-major), K = {half*8..+7} U {16+half*8..+7}
__device__ inline v16bf ld_bfrag(const u16* rowbase, int kb, int half) {
  v8bf lo = *(const v8bf*)(rowbase + kb + half * 8);
  v8bf hi = *(const v8bf*)(rowbase + kb + 16 + half * 8);
  return __builtin_shufflevector(lo, hi, 0,1,2,3,4,5,6,7,8,9,10,11,12,13,14,15);
}
__device__ inline float sigm(float x) { return 1.f / (1.f + __expf(-x)); }

// LDS byte offset of a generic pointer to __shared__ memory: flat LDS aperture
// keeps the LDS offset in addr[31:0] (ISA 10.2 aperture rules).
__device__ inline unsigned lds_off(const void* p) {
  return (unsigned)(unsigned long long)p;
}
// CDNA5 async copy: 16B per lane, global -> LDS, tracked by ASYNCcnt (no VGPR round-trip)
__device__ inline void async_b128(unsigned lds_byte_addr, const void* gaddr) {
  asm volatile("global_load_async_to_lds_b128 %0, %1, off"
               :: "v"(lds_byte_addr), "v"(gaddr) : "memory");
}
__device__ inline void wait_async0() { asm volatile("s_wait_asynccnt 0x0" ::: "memory"); }
__device__ inline void wait_async3() { asm volatile("s_wait_asynccnt 0x3" ::: "memory"); }

// CDNA5 Tensor Data Mover: 2D tile (rows x cols elements of 2 bytes) global -> LDS.
// D# per ISA ch.8: group0 {count=1, lds_addr, global_addr, type=2},
// group1 {data_size=2B, tensor_dim0/1, tile_dim0/1, tensor_dim0_stride}; groups 2/3 zero (2D).
// This toolchain exposes the 6-arg builtin: (v4u, v8i, v4i, v4i, v8i, i32 cpol).
__device__ inline void tdm_load_2d_bf16(unsigned lds_byte_addr, const void* gaddr,
                                        unsigned rows, unsigned cols, unsigned stride_elems) {
  unsigned long long ga = (unsigned long long)gaddr;
  v4u g0;
  g0[0] = 1u;                                           // count=1, user descriptor
  g0[1] = lds_byte_addr;                                // D#.lds_addr
  g0[2] = (unsigned)(ga & 0xFFFFFFFFu);                 // global_addr[31:0]
  g0[3] = (unsigned)((ga >> 32) & 0x01FFFFFFu) | (2u << 30);  // global_addr[56:32], type=2
  v8i g1;
  g1[0] = (int)(1u << 16);                              // data_size=1 (2 bytes); mask=0
  g1[1] = (int)((cols & 0xFFFFu) << 16);                // tensor_dim0[15:0]
  g1[2] = (int)(((cols >> 16) & 0xFFFFu) | ((rows & 0xFFFFu) << 16)); // dim0 hi | tensor_dim1 lo
  g1[3] = (int)(((rows >> 16) & 0xFFFFu) | ((cols & 0xFFFFu) << 16)); // dim1 hi | tile_dim0
  g1[4] = (int)(rows & 0xFFFFu);                        // tile_dim1; tile_dim2=0
  g1[5] = (int)stride_elems;                            // tensor_dim0_stride[31:0]
  g1[6] = 0;                                            // stride hi | dim1_stride lo
  g1[7] = 0;
  v4i z4 = {0, 0, 0, 0};
  v8i z8 = {0, 0, 0, 0, 0, 0, 0, 0};
  __builtin_amdgcn_tensor_load_to_lds(g0, g1, z4, z4, z8, 0);
}

// ---------------- tiny utility kernels ----------------
__global__ void k_zero(unsigned* p, int n) {
  for (int i = blockIdx.x * blockDim.x + threadIdx.x; i < n; i += gridDim.x * blockDim.x) p[i] = 0u;
}

// fp32 -> bf16 with optional zero-padded K (column) dimension
__global__ void k_f2bf_pad(const float* __restrict__ src, u16* __restrict__ dst,
                           int rows, int scols, int dcols) {
  long long n = (long long)rows * dcols;
  for (long long i = blockIdx.x * (long long)blockDim.x + threadIdx.x; i < n;
       i += (long long)gridDim.x * blockDim.x) {
    int r = (int)(i / dcols), c = (int)(i % dcols);
    dst[i] = (c < scols) ? f2bf(src[(long long)r * scols + c]) : (u16)0;
  }
}

// gather embedding rows into seq-major bf16 A matrix (4096 x 320), zero-padded K
__global__ void k_embed(const int* __restrict__ x, const float* __restrict__ emb,
                        u16* __restrict__ X0) {
  int n = TOK * DP_;
  for (int i = blockIdx.x * blockDim.x + threadIdx.x; i < n; i += gridDim.x * blockDim.x) {
    int m = i / DP_, k = i % DP_;
    int b = m & 31, s = m >> 5;            // token m = s*32 + b
    int id = x[b * S_ + s];
    X0[i] = (k < D_) ? f2bf(emb[(long long)id * D_ + k]) : (u16)0;
  }
}

// ---------------- generic WMMA GEMM: C[M,N](bf16) = A(MxK bf16) @ W(NxK bf16)^T + bias ----
// Double-buffered LDS tiles filled by async global->LDS copies (ASYNCcnt pipeline).
__global__ void __launch_bounds__(256) k_gemm(const u16* __restrict__ A, int lda,
                                              const u16* __restrict__ W, int ldw,
                                              const float* __restrict__ bias,
                                              u16* __restrict__ C, int ldc, int K) {
  __shared__ __align__(32) u16 Asm[2][128][32];
  __shared__ __align__(32) u16 Wsm[2][64][32];
  const int tid = threadIdx.x;
  const int lane = tid & 31, wave = tid >> 5;
  const int r = lane & 15, half = lane >> 4;
  const int wm = wave & 3, wn = wave >> 2;          // 4x2 wave grid, each wave 32x32
  const int row0 = blockIdx.y * 128, col0 = blockIdx.x * 64;

  // stage one 128x32 A tile + 64x32 W tile into LDS buffer `bf` (3 async instrs/wave)
  auto stage = [&](int kb, int bf) {
    #pragma unroll
    for (int i = 0; i < 2; i++) {
      int ch = tid + 256 * i;
      int rr = ch >> 2, seg = ch & 3;
      async_b128(lds_off(&Asm[bf][rr][seg * 8]),
                 &A[(long long)(row0 + rr) * lda + kb + seg * 8]);
    }
    int rr = tid >> 2, seg = tid & 3;
    async_b128(lds_off(&Wsm[bf][rr][seg * 8]),
               &W[(long long)(col0 + rr) * ldw + kb + seg * 8]);
  };

  v8f acc[2][2];
  for (int nt = 0; nt < 2; nt++) {
    float bv = bias[col0 + wn * 32 + nt * 16 + r];
    for (int mt = 0; mt < 2; mt++)
      for (int j = 0; j < 8; j++) acc[mt][nt][j] = bv;
  }

  stage(0, 0);
  int buf = 0;
  for (int kb = 0; kb < K; kb += 32, buf ^= 1) {
    const bool nxt = (kb + 32) < K;
    if (nxt) stage(kb + 32, buf ^ 1);
    if (nxt) wait_async3();           // allow next stage's 3 ops to stay in flight
    else     wait_async0();
    __syncthreads();                  // all waves' async copies for `buf` landed
    v16bf a[2], b[2];
    #pragma unroll
    for (int mt = 0; mt < 2; mt++) a[mt] = ld_afrag(&Asm[buf][wm * 32 + mt * 16 + r][0], 0, half);
    #pragma unroll
    for (int nt = 0; nt < 2; nt++) b[nt] = ld_bfrag(&Wsm[buf][wn * 32 + nt * 16 + r][0], 0, half);
    __syncthreads();                  // frags in VGPRs; `buf` may be overwritten next iter
    #pragma unroll
    for (int mt = 0; mt < 2; mt++)
      #pragma unroll
      for (int nt = 0; nt < 2; nt++) acc[mt][nt] = wmma_bf16(a[mt], b[nt], acc[mt][nt]);
  }
  for (int mt = 0; mt < 2; mt++)
    for (int nt = 0; nt < 2; nt++) {
      int col = col0 + wn * 32 + nt * 16 + r;
      #pragma unroll
      for (int j = 0; j < 8; j++) {
        int row = row0 + wm * 32 + mt * 16 + j + half * 8;   // D: M = j + 8*half
        C[(long long)row * ldc + col] = f2bf(acc[mt][nt][j]);
      }
    }
}

// ---------------- persistent BiLSTM recurrence ----------------
// grid = (32 unit-chunks, 2 directions), block = 256 (8 waves)
// Each WG owns 32 hidden units: gate columns {g*1024 + u0..u0+31}; its 128x1024 bf16
// Whh slice is DMA'd into LDS once by the Tensor Data Mover (4 x 2D tile loads,
// TENSORcnt) and reused for all 128 steps. Cell state c lives in registers. h state
// (32x1024 bf16 per dir) is double-buffered in global; per-step cross-WG barrier
// uses one counter per (dir,step).
__global__ void __launch_bounds__(256) k_lstm(const u16* __restrict__ pre0,
                                              const u16* __restrict__ pre1,
                                              const u16* __restrict__ whh,    // [2][4096][1024] bf16
                                              u16* __restrict__ hs_out,       // [4096][2048] bf16
                                              u16* __restrict__ hstate,       // [2 dir][2 buf][32][1024]
                                              unsigned* __restrict__ cnt) {   // [2][128]
  extern __shared__ unsigned char smem[];
  u16*  whh_lds = (u16*)smem;                                   // [128][1024] bf16 = 256KB
  float* gst    = (float*)(smem + 128 * H_ * sizeof(u16));      // [32][128] f32 = 16KB

  const int tid = threadIdx.x;
  const int lane = tid & 31, wave = tid >> 5;
  const int r = lane & 15, half = lane >> 4;
  const int dir = blockIdx.y;
  const int u0 = blockIdx.x * 32;
  const int mt = wave & 1;               // 2 M-tiles (batch 32)
  const int nt0 = (wave >> 1) * 2;       // 8 N-tiles of 16 local gate cols; 2 per wave
  const u16* pre = dir ? pre1 : pre0;
  const u16* whh_d = whh + (long long)dir * G4H * H_;

  // TDM preload of Whh slice: local rows c = g*32+uu <- global rows g*1024+u0+uu.
  // One 2D tile descriptor per gate block: 32 rows x 1024 cols bf16, row stride 1024.
  if (wave == 0) {
    for (int g = 0; g < 4; ++g) {
      tdm_load_2d_bf16(lds_off(whh_lds) + (unsigned)(g * 32 * H_ * 2),
                       whh_d + (long long)(g * H_ + u0) * H_,
                       32u, (unsigned)H_, (unsigned)H_);
    }
  }
  __builtin_amdgcn_s_wait_tensorcnt(0);
  __syncthreads();

  float c_reg[4] = {0.f, 0.f, 0.f, 0.f}; // persistent cell state, 4 (b,uu) pairs per thread

  for (int s = 0; s < S_; ++s) {
    const int t = dir ? (S_ - 1 - s) : s;
    const u16* hstR = hstate + (dir * 2 + ((s + 1) & 1)) * B_ * H_;  // written at step s-1
    u16*       hstW = hstate + (dir * 2 + (s & 1)) * B_ * H_;

    // init accumulators from precomputed input projection
    v8f acc[2];
    for (int q = 0; q < 2; q++) {
      int c = (nt0 + q) * 16 + r;                       // local gate col 0..127
      int ng = (c >> 5) * H_ + u0 + (c & 31);           // global gate col
      #pragma unroll
      for (int j = 0; j < 8; j++) {
        int b = mt * 16 + j + half * 8;
        acc[q][j] = bf2f(pre[(long long)(t * B_ + b) * G4H + ng]);
      }
    }
    // prefetch next step's pre slice into cache while we do the WMMA work
    if (s + 1 < S_) {
      const int tn = dir ? (S_ - 2 - s) : (s + 1);
      __builtin_prefetch(&pre[(long long)(tn * B_ + (tid & 31)) * G4H + u0], 0, 0);
    }
    if (s > 0) {                                        // + h_{s-1} @ Whh^T
      for (int kb = 0; kb < H_; kb += 32) {
        v16bf a = ld_afrag(&hstR[(mt * 16 + r) * H_], kb, half);
        #pragma unroll
        for (int q = 0; q < 2; q++) {
          int c = (nt0 + q) * 16 + r;
          v16bf b = ld_bfrag(&whh_lds[c * H_], kb, half);
          acc[q] = wmma_bf16(a, b, acc[q]);
        }
      }
    }
    // stage gates to LDS so each thread can combine i,f,g,o
    for (int q = 0; q < 2; q++) {
      int c = (nt0 + q) * 16 + r;
      #pragma unroll
      for (int j = 0; j < 8; j++) {
        int b = mt * 16 + j + half * 8;
        gst[b * 128 + c] = acc[q][j];
      }
    }
    __syncthreads();
    // elementwise LSTM cell
    #pragma unroll
    for (int q = 0; q < 4; q++) {
      int p = tid + 256 * q;
      int b = p >> 5, uu = p & 31;
      float gi = gst[b * 128 + uu];
      float gf = gst[b * 128 + 32 + uu];
      float gg = gst[b * 128 + 64 + uu];
      float go = gst[b * 128 + 96 + uu];
      float cc = sigm(gf) * c_reg[q] + sigm(gi) * tanhf(gg);
      c_reg[q] = cc;
      u16 hb = f2bf(sigm(go) * tanhf(cc));
      hstW[b * H_ + u0 + uu] = hb;
      hs_out[(long long)(t * B_ + b) * H2_ + dir * H_ + u0 + uu] = hb;
    }
    if (s < S_ - 1) {                                   // cross-WG step barrier (per dir, per step)
      __threadfence();
      __syncthreads();
      unsigned* cp = cnt + dir * S_ + s;
      if (tid == 0) {
        atomicAdd(cp, 1u);
        volatile unsigned* vp = cp;
        while (*vp < gridDim.x) __builtin_amdgcn_s_sleep(1);
      }
      __syncthreads();
      __threadfence();
    }
  }
}

// ---------------- BN(eval)+ReLU+linear head: emis[b][s][9] ----------------
__global__ void __launch_bounds__(256) k_head(const u16* __restrict__ hs,
                                              const float* __restrict__ gamma,
                                              const float* __restrict__ beta,
                                              const float* __restrict__ w2,
                                              const float* __restrict__ b2,
                                              float* __restrict__ emis) {
  __shared__ float red[256][NT_];
  const int m = blockIdx.x;            // token = s*32 + b
  const int tid = threadIdx.x;
  const float inv = rsqrtf(1.0f + 1e-5f);
  float p[NT_];
  for (int t = 0; t < NT_; t++) p[t] = 0.f;
  for (int k = tid; k < H2_; k += 256) {
    float hv = bf2f(hs[(long long)m * H2_ + k]);
    hv = gamma[k] * hv * inv + beta[k];
    hv = hv > 0.f ? hv : 0.f;
    for (int t = 0; t < NT_; t++) p[t] += hv * w2[t * H2_ + k];
  }
  for (int t = 0; t < NT_; t++) red[tid][t] = p[t];
  __syncthreads();
  for (int off = 128; off > 0; off >>= 1) {
    if (tid < off)
      for (int t = 0; t < NT_; t++) red[tid][t] += red[tid + off][t];
    __syncthreads();
  }
  if (tid < NT_) {
    int b = m & 31, s = m >> 5;
    emis[((long long)b * S_ + s) * NT_ + tid] = red[0][tid] + b2[tid];
  }
}

// ---------------- CRF log-likelihood (one wave, lane = batch) ----------------
__global__ void k_crf(const float* __restrict__ emis, const int* __restrict__ y,
                      const unsigned char* __restrict__ mask,
                      const float* __restrict__ start, const float* __restrict__ endv,
                      const float* __restrict__ trans, float* __restrict__ out) {
  __shared__ float tr[NT_ * NT_], st[NT_], en[NT_];
  int tid = threadIdx.x;
  if (tid < NT_ * NT_) tr[tid] = trans[tid];
  if (tid < NT_) { st[tid] = start[tid]; en[tid] = endv[tid]; }
  __syncthreads();
  if (tid < B_) {
    int b = tid;
    const float* e = emis + (long long)b * S_ * NT_;
    const int* tg = y + b * S_;
    const unsigned char* mk = mask + b * S_;

    float alpha[NT_];
    int t0 = tg[0];
    float num = st[t0] + e[t0];
    for (int k = 0; k < NT_; k++) alpha[k] = st[k] + e[k];
    int prev = t0;
    int len = mk[0] ? 1 : 0;
    for (int s = 1; s < S_; s++) {
      int ts = tg[s];
      float mf = mk[s] ? 1.f : 0.f;
      num += mf * (tr[prev * NT_ + ts] + e[s * NT_ + ts]);
      prev = ts;
      if (mk[s]) len++;
      float na[NT_];
      for (int j = 0; j < NT_; j++) {
        float mx = -3.4e38f;
        for (int i = 0; i < NT_; i++) mx = fmaxf(mx, alpha[i] + tr[i * NT_ + j]);
        float ss = 0.f;
        for (int i = 0; i < NT_; i++) ss += __expf(alpha[i] + tr[i * NT_ + j] - mx);
        na[j] = mx + __logf(ss) + e[s * NT_ + j];
      }
      if (mk[s]) for (int j = 0; j < NT_; j++) alpha[j] = na[j];
    }
    int li = (len > 0) ? (len - 1) : 0;
    num += en[tg[li]];
    float mx = -3.4e38f;
    for (int k = 0; k < NT_; k++) mx = fmaxf(mx, alpha[k] + en[k]);
    float ss = 0.f;
    for (int k = 0; k < NT_; k++) ss += __expf(alpha[k] + en[k] - mx);
    float val = num - (mx + __logf(ss));
    for (int off = 16; off > 0; off >>= 1) val += __shfl_xor(val, off, 32);
    if (tid == 0) out[0] = val;
  }
}

// ---------------- host orchestration ----------------
extern "C" void kernel_launch(void* const* d_in, const int* in_sizes, int n_in,
                              void* d_out, int out_size, void* d_ws, size_t ws_size,
                              hipStream_t stream) {
  (void)in_sizes; (void)n_in; (void)out_size; (void)ws_size;
  const int* x = (const int*)d_in[0];
  const int* y = (const int*)d_in[1];
  const unsigned char* mask = (const unsigned char*)d_in[2];
  const float* emb    = (const float*)d_in[3];
  const float* lin1_w = (const float*)d_in[4];
  const float* lin1_b = (const float*)d_in[5];
  const float* wih[4]  = {(const float*)d_in[6], (const float*)d_in[9],
                          (const float*)d_in[12], (const float*)d_in[15]};
  const float* whh[4]  = {(const float*)d_in[7], (const float*)d_in[10],
                          (const float*)d_in[13], (const float*)d_in[16]};
  const float* bias[4] = {(const float*)d_in[8], (const float*)d_in[11],
                          (const float*)d_in[14], (const float*)d_in[17]};
  const float* gamma = (const float*)d_in[18];
  const float* beta  = (const float*)d_in[19];
  const float* w2    = (const float*)d_in[20];
  const float* b2    = (const float*)d_in[21];
  const float* cs    = (const float*)d_in[22];
  const float* ce    = (const float*)d_in[23];
  const float* ct    = (const float*)d_in[24];
  float* out = (float*)d_out;

  unsigned char* ws = (unsigned char*)d_ws;
  size_t off = 0;
  auto alloc = [&](size_t bytes) -> unsigned char* {
    unsigned char* p = ws + off;
    off += (bytes + 255) & ~(size_t)255;
    return p;
  };
  u16* l1w_bf = (u16*)alloc((size_t)512 * DP_ * 2);
  int insz[4] = {512, H2_, H2_, H2_};
  u16* wih_bf[4]; u16* whh_bf[4];
  for (int l = 0; l < 4; l++) {
    wih_bf[l] = (u16*)alloc((size_t)2 * G4H * insz[l] * 2);
    whh_bf[l] = (u16*)alloc((size_t)2 * G4H * H_ * 2);
  }
  u16* X0  = (u16*)alloc((size_t)TOK * DP_ * 2);
  u16* Hin = (u16*)alloc((size_t)TOK * 512 * 2);
  u16* HA  = (u16*)alloc((size_t)TOK * H2_ * 2);
  u16* HB  = (u16*)alloc((size_t)TOK * H2_ * 2);
  u16* PRE0 = (u16*)alloc((size_t)TOK * G4H * 2);
  u16* PRE1 = (u16*)alloc((size_t)TOK * G4H * 2);
  u16* HST  = (u16*)alloc((size_t)2 * 2 * B_ * H_ * 2);   // [dir][buf][32][1024]
  unsigned* CNT = (unsigned*)alloc((size_t)4 * 2 * S_ * 4);
  float* EMIS = (float*)alloc((size_t)B_ * S_ * NT_ * 4);

  k_zero<<<4, 256, 0, stream>>>(CNT, 4 * 2 * S_);
  k_f2bf_pad<<<256, 256, 0, stream>>>(lin1_w, l1w_bf, 512, D_, DP_);
  for (int l = 0; l < 4; l++) {
    k_f2bf_pad<<<4096, 256, 0, stream>>>(wih[l], wih_bf[l], 2 * G4H, insz[l], insz[l]);
    k_f2bf_pad<<<4096, 256, 0, stream>>>(whh[l], whh_bf[l], 2 * G4H, H_, H_);
  }
  k_embed<<<2048, 256, 0, stream>>>(x, emb, X0);

  // lin1: (4096x320) @ (512x320)^T
  k_gemm<<<dim3(512 / 64, TOK / 128), 256, 0, stream>>>(X0, DP_, l1w_bf, DP_, lin1_b,
                                                        Hin, 512, DP_);

  const u16* cur = Hin; int curK = 512;
  u16* outs[4] = {HA, HB, HA, HB};
  size_t lstm_lds = (size_t)128 * H_ * sizeof(u16) + (size_t)B_ * 128 * sizeof(float); // 272KB
  for (int l = 0; l < 4; l++) {
    for (int d = 0; d < 2; ++d) {
      k_gemm<<<dim3(G4H / 64, TOK / 128), 256, 0, stream>>>(
          cur, curK, wih_bf[l] + (size_t)d * G4H * curK, curK,
          bias[l] + d * G4H, d ? PRE1 : PRE0, G4H, curK);
    }
    k_lstm<<<dim3(32, 2), 256, lstm_lds, stream>>>(PRE0, PRE1, whh_bf[l], outs[l], HST,
                                                   CNT + l * 2 * S_);
    cur = outs[l]; curK = H2_;
  }
  k_head<<<TOK, 256, 0, stream>>>(cur, gamma, beta, w2, b2, EMIS);
  k_crf<<<1, 96, 0, stream>>>(EMIS, y, mask, cs, ce, ct, out);
}
